// deform_fusion_40183714021976
// MI455X (gfx1250) — matrix operand order repaired and checked by reading, not compiled, not run on hardware
//
#include <hip/hip_runtime.h>
#include <stdint.h>

typedef __attribute__((ext_vector_type(16))) __bf16 v16bf;
typedef __attribute__((ext_vector_type(8)))  __bf16 v8bf;
typedef __attribute__((ext_vector_type(8)))  float  v8f;

#define AS1 __attribute__((address_space(1)))
typedef const AS1 __bf16* gcbf;   // global const bf16*
typedef AS1 __bf16*       gbf;    // global bf16*
typedef const AS1 float*  gcf;    // global const float*
typedef AS1 float*        gf;     // global float*

union FragBF {
  v16bf v;
  v8bf  h[2];
  uint4 u[2];
};

__device__ __forceinline__ int iclampi(int v, int lo, int hi) {
  return v < lo ? lo : (v > hi ? hi : v);
}

// ---------------- Pack kernels ----------------

__global__ void zero_init_bf16(__bf16* __restrict__ z, int n) {
  int i = blockIdx.x * blockDim.x + threadIdx.x;
  if (i < n) z[i] = (__bf16)0.f;
}

// NCHW f32 -> NHWC bf16
__global__ void nchw2nhwc_bf16(const float* __restrict__ in, __bf16* __restrict__ out,
                               int B, int C, int H, int W) {
  int idx = blockIdx.x * blockDim.x + threadIdx.x;
  int total = B * C * H * W;
  if (idx >= total) return;
  int c = idx % C;
  int t = idx / C;
  int x = t % W; t /= W;
  int y = t % H; t /= H;
  int b = t;
  float v = in[(((size_t)b * C + c) * H + y) * W + x];
  out[idx] = (__bf16)v;
}

// NCHW f32 -> NHWC f32
__global__ void nchw2nhwc_f32(const float* __restrict__ in, float* __restrict__ out,
                              int B, int C, int H, int W) {
  int idx = blockIdx.x * blockDim.x + threadIdx.x;
  int total = B * C * H * W;
  if (idx >= total) return;
  int c = idx % C;
  int t = idx / C;
  int x = t % W; t /= W;
  int y = t % H; t /= H;
  int b = t;
  out[idx] = in[(((size_t)b * C + c) * H + y) * W + x];
}

// [O,C,3,3] f32 -> [Opad, 9, C] bf16 (K = k*C + c), zero-padded rows
__global__ void pack_weights(const float* __restrict__ w, __bf16* __restrict__ out,
                             int O, int Opad, int C) {
  int idx = blockIdx.x * blockDim.x + threadIdx.x;
  int total = Opad * 9 * C;
  if (idx >= total) return;
  int c = idx % C;
  int k = (idx / C) % 9;
  int o = idx / (9 * C);
  float v = (o < O) ? w[((size_t)o * C + c) * 9 + k] : 0.f;
  out[idx] = (__bf16)v;
}

// ---------------- Deform bilinear sampling (f32 math) ----------------
// Produces im2col matrix sbuf[(b*784 + n)][k*768 + c] in bf16.
__global__ void deform_sample(const float* __restrict__ cnnhwc,   // [8,28,28,768] f32
                              const float* __restrict__ off,       // [8,18,784] f32
                              __bf16* __restrict__ sbuf) {
  const int HW = 784, W = 28, C = 768;
  int bn = blockIdx.x;
  int b = bn / HW, n = bn % HW;
  int y = n / W, x = n % W;
  int k = blockIdx.y;

  float ody = off[((size_t)b * 18 + 2 * k) * HW + n];
  float odx = off[((size_t)b * 18 + 2 * k + 1) * HW + n];
  float py = (float)(y + (k / 3) - 1) + ody;
  float px = (float)(x + (k % 3) - 1) + odx;
  float y0f = floorf(py), x0f = floorf(px);
  float ly = py - y0f, lx = px - x0f;
  int y0 = (int)y0f, x0 = (int)x0f;
  int y1 = y0 + 1, x1 = x0 + 1;

  float w00 = (1.f - ly) * (1.f - lx);
  float w01 = (1.f - ly) * lx;
  float w10 = ly * (1.f - lx);
  float w11 = ly * lx;
  if (!(y0 >= 0 && y0 < 28 && x0 >= 0 && x0 < 28)) w00 = 0.f;
  if (!(y0 >= 0 && y0 < 28 && x1 >= 0 && x1 < 28)) w01 = 0.f;
  if (!(y1 >= 0 && y1 < 28 && x0 >= 0 && x0 < 28)) w10 = 0.f;
  if (!(y1 >= 0 && y1 < 28 && x1 >= 0 && x1 < 28)) w11 = 0.f;

  int cy0 = iclampi(y0, 0, 27), cy1 = iclampi(y1, 0, 27);
  int cx0 = iclampi(x0, 0, 27), cx1 = iclampi(x1, 0, 27);

  const float* base = cnnhwc + (size_t)b * HW * C;
  const float* p00 = base + ((size_t)cy0 * 28 + cx0) * C;
  const float* p01 = base + ((size_t)cy0 * 28 + cx1) * C;
  const float* p10 = base + ((size_t)cy1 * 28 + cx0) * C;
  const float* p11 = base + ((size_t)cy1 * 28 + cx1) * C;

  __bf16* so = sbuf + (size_t)bn * 6912 + (size_t)k * C;
  for (int c = threadIdx.x; c < C; c += blockDim.x) {
    float a = w00 * p00[c] + w01 * p01[c] + w10 * p10[c] + w11 * p11[c];
    so[c] = (__bf16)a;
  }
}

// ---------------- Generic implicit-GEMM conv with bf16 WMMA ----------------
// D[m,n] = sum_K A[m,K] * Bmat[n,K]  (+bias, opt relu)
// BMODE 0: direct [N, Ktot] bf16 matrix (1 tap of length Ktot)
// BMODE 1: implicit 3x3 pad-1 im2col from NHWC bf16 `act`, K = tap*Cin + c.
//          Per tap: compute bounds once; OOB lanes read a zero buffer with step 0.
// All hot pointers are explicit addrspace(1) so loads lower to global_load_b128
// (not flat_load: FLAT couples LOADcnt with DScnt and burns the LDS port).
#define LOAD_FRAGS(AF, BF)                                   \
  _Pragma("unroll") for (int mt = 0; mt < MT; ++mt) {        \
    AF[mt].h[0] = *(const AS1 v8bf*)ap[mt];                  \
    AF[mt].h[1] = *(const AS1 v8bf*)(ap[mt] + 8);            \
    ap[mt] += 32;                                            \
  }                                                          \
  _Pragma("unroll") for (int nt = 0; nt < 2; ++nt) {         \
    BF[nt].v = *(const AS1 v16bf*)bp[nt];                    \
    bp[nt] += bstep[nt];                                     \
  }

#define DO_WMMA(AF, BF)                                      \
  _Pragma("unroll") for (int mt = 0; mt < MT; ++mt)          \
    _Pragma("unroll") for (int nt = 0; nt < 2; ++nt)         \
      acc[mt][nt] = __builtin_amdgcn_wmma_f32_16x16x32_bf16( \
          false, AF[mt].v, false, BF[nt].v, (short)0, acc[mt][nt], false, false);

template <int BMODE, int MT>
__global__ void __launch_bounds__(128)
wmma_conv_gemm(const __bf16* __restrict__ A_, int Ktot, int Mreal,
               const __bf16* __restrict__ Bdir_,
               const __bf16* __restrict__ act_,
               const __bf16* __restrict__ zbuf_,
               int Hin, int Win, int Cin, int Hb, int Wb,
               int Hout, int Wout, int stride, int srcShift,
               const float* __restrict__ bias_,
               float* __restrict__ outF_, __bf16* __restrict__ outB_,
               int outMode, int doRelu, int Ntot) {
  // promote to explicit global address space
  gcbf A    = (gcbf)A_;
  gcbf Bdir = (gcbf)Bdir_;
  gcbf act  = (gcbf)act_;
  gcbf zbuf = (gcbf)zbuf_;
  gcf  bias = (gcf)bias_;
  gf   outF = (gf)outF_;
  gbf  outB = (gbf)outB_;

  int lane = threadIdx.x & 31;
  int wv = threadIdx.x >> 5;
  int n0 = (blockIdx.x * 4 + wv) * 32;
  if (n0 >= Ntot) return;              // wave-uniform: EXEC stays all-ones for WMMA
  int m0 = blockIdx.y * (MT * 16);
  int half = lane >> 4;
  int ln = lane & 15;
  int HWout = Hout * Wout;

  // per-lane column decomposition for the two 16-wide N tiles
  int nIdx[2], nBb[2], nY[2], nX[2];
#pragma unroll
  for (int nt = 0; nt < 2; ++nt) {
    int n = n0 + nt * 16 + ln;
    nIdx[nt] = n;
    int bb = n / HWout;
    int p = n - bb * HWout;
    int yy = p / Wout;
    nBb[nt] = bb; nY[nt] = yy; nX[nt] = p - yy * Wout;
  }

  v8f zero = {0.f, 0.f, 0.f, 0.f, 0.f, 0.f, 0.f, 0.f};
  v8f acc[MT][2];
#pragma unroll
  for (int i = 0; i < MT; ++i)
#pragma unroll
    for (int j = 0; j < 2; ++j) acc[i][j] = zero;

  // A row pointers: K is contiguous across all taps, advance monotonically.
  gcbf ap[MT];
#pragma unroll
  for (int mt = 0; mt < MT; ++mt)
    ap[mt] = A + (size_t)(m0 + mt * 16 + ln) * Ktot + half * 16;

  const int nTaps = (BMODE == 0) ? 1 : 9;
  const int tapLen = (BMODE == 0) ? Ktot : Cin;

  for (int tap = 0; tap < nTaps; ++tap) {
    // ---- per-tap B setup (hoisted out of the hot loop) ----
    gcbf bp[2];
    int bstep[2];
    if (BMODE == 0) {
#pragma unroll
      for (int nt = 0; nt < 2; ++nt) {
        bp[nt] = Bdir + (size_t)nIdx[nt] * Ktot + half * 16;
        bstep[nt] = 32;
      }
    } else {
      int dy = tap / 3 - 1;
      int dx = tap % 3 - 1;
#pragma unroll
      for (int nt = 0; nt < 2; ++nt) {
        int sy = stride * nY[nt] + dy;
        int sx = stride * nX[nt] + dx;
        bool ok = (sy >= 0) & (sy < Hb) & (sx >= 0) & (sx < Wb);
        bp[nt] = ok ? act + ((size_t)(nBb[nt] * Hin + (sy >> srcShift)) * Win +
                             (sx >> srcShift)) * Cin + half * 16
                    : zbuf;
        bstep[nt] = ok ? 32 : 0;
      }
    }
    // ---- 2-stage software-pipelined inner loop (tapLen/32 is even for all uses) ----
    FragBF a0[MT], b0[2];
    LOAD_FRAGS(a0, b0)
    for (int cb = 32; cb < tapLen; cb += 32) {
      FragBF a1[MT], b1[2];
      LOAD_FRAGS(a1, b1)
      DO_WMMA(a0, b0)
#pragma unroll
      for (int mt = 0; mt < MT; ++mt) a0[mt].v = a1[mt].v;
#pragma unroll
      for (int nt = 0; nt < 2; ++nt) b0[nt].v = b1[nt].v;
    }
    DO_WMMA(a0, b0)
  }

  // ---- Epilogue: C/D layout VGPR r -> M = r + 8*half ----
#pragma unroll
  for (int mt = 0; mt < MT; ++mt) {
#pragma unroll
    for (int nt = 0; nt < 2; ++nt) {
#pragma unroll
      for (int r = 0; r < 8; ++r) {
        int m = m0 + mt * 16 + r + 8 * half;
        if (m >= Mreal) continue;
        float v = acc[mt][nt][r] + bias[m];
        if (doRelu) v = fmaxf(v, 0.f);
        if (outMode == 0) {
          outF[((size_t)nBb[nt] * Mreal + m) * HWout + (nY[nt] * Wout + nX[nt])] = v;
        } else {
          outB[(size_t)nIdx[nt] * Mreal + m] = (__bf16)v;
        }
      }
    }
  }
}

// ---------------- Host launcher ----------------
extern "C" void kernel_launch(void* const* d_in, const int* in_sizes, int n_in,
                              void* d_out, int out_size, void* d_ws, size_t ws_size,
                              hipStream_t stream) {
  const float* cnn   = (const float*)d_in[0];  // [8,768,28,28]
  const float* vit   = (const float*)d_in[1];  // [8,3840,14,14]
  const float* w_off = (const float*)d_in[2];  // [18,3840,3,3]
  const float* b_off = (const float*)d_in[3];
  const float* w_def = (const float*)d_in[4];  // [768,768,3,3]
  const float* b_def = (const float*)d_in[5];
  const float* w1    = (const float*)d_in[6];  // [512,768,3,3]
  const float* b1    = (const float*)d_in[7];
  const float* w2    = (const float*)d_in[8];  // [768,512,3,3]
  const float* b2    = (const float*)d_in[9];
  float* out = (float*)d_out;                  // [8,768,14,14]

  char* p = (char*)d_ws;
  auto alloc = [&](size_t bytes) -> char* {
    char* r = p;
    p += (bytes + 255) & ~(size_t)255;
    return r;
  };
  __bf16* vhwc = (__bf16*)alloc((size_t)8 * 14 * 14 * 3840 * 2); // vit NHWC bf16
  float*  chwc = (float*) alloc((size_t)8 * 28 * 28 * 768 * 4);  // cnn NHWC f32
  __bf16* Aoff = (__bf16*)alloc((size_t)32  * 34560 * 2);
  __bf16* Adef = (__bf16*)alloc((size_t)768 * 6912 * 2);
  __bf16* Aw1  = (__bf16*)alloc((size_t)512 * 6912 * 2);
  __bf16* Aw2  = (__bf16*)alloc((size_t)768 * 4608 * 2);
  float*  offb = (float*) alloc((size_t)8 * 18 * 784 * 4);       // offsets f32
  __bf16* sbuf = (__bf16*)alloc((size_t)6272 * 6912 * 2);        // deform im2col
  __bf16* dbuf = (__bf16*)alloc((size_t)8 * 784 * 768 * 2);      // d NHWC bf16
  __bf16* hbuf = (__bf16*)alloc((size_t)8 * 196 * 512 * 2);      // h NHWC bf16
  __bf16* zbuf = (__bf16*)alloc((size_t)256);                    // zeros for OOB lanes

  zero_init_bf16<<<1, 128, 0, stream>>>(zbuf, 128);
  { int tot = 8 * 3840 * 196; nchw2nhwc_bf16<<<(tot + 255) / 256, 256, 0, stream>>>(vit, vhwc, 8, 3840, 14, 14); }
  { int tot = 8 * 768 * 784;  nchw2nhwc_f32 <<<(tot + 255) / 256, 256, 0, stream>>>(cnn, chwc, 8, 768, 28, 28); }
  { int tot = 32 * 9 * 3840;  pack_weights<<<(tot + 255) / 256, 256, 0, stream>>>(w_off, Aoff, 18, 32, 3840); }
  { int tot = 768 * 9 * 768;  pack_weights<<<(tot + 255) / 256, 256, 0, stream>>>(w_def, Adef, 768, 768, 768); }
  { int tot = 512 * 9 * 768;  pack_weights<<<(tot + 255) / 256, 256, 0, stream>>>(w1, Aw1, 512, 512, 768); }
  { int tot = 768 * 9 * 512;  pack_weights<<<(tot + 255) / 256, 256, 0, stream>>>(w2, Aw2, 768, 768, 512); }

  dim3 blk(128);
  // GEMM1: offset conv (nearest-resize fused via srcShift=1). M=18(pad32) K=34560 N=6272
  wmma_conv_gemm<1, 2><<<dim3(49, 1), blk, 0, stream>>>(
      Aoff, 34560, 18, nullptr, vhwc, zbuf,
      14, 14, 3840, 28, 28, 28, 28, 1, 1,
      b_off, offb, nullptr, /*outMode*/0, /*relu*/0, 6272);

  // bilinear deform sampling -> sbuf
  deform_sample<<<dim3(8 * 784, 9), 256, 0, stream>>>(chwc, offb, sbuf);

  // GEMM2: deform conv einsum. M=768 K=6912 N=6272, direct B
  wmma_conv_gemm<0, 4><<<dim3(49, 12), blk, 0, stream>>>(
      Adef, 6912, 768, sbuf, nullptr, zbuf,
      0, 0, 1, 0, 0, 28, 28, 1, 0,
      b_def, nullptr, dbuf, /*outMode*/1, /*relu*/0, 6272);

  // GEMM3: stride-2 conv + relu. M=512 K=6912 N=1568
  wmma_conv_gemm<1, 4><<<dim3(13, 8), blk, 0, stream>>>(
      Aw1, 6912, 512, nullptr, dbuf, zbuf,
      28, 28, 768, 28, 28, 14, 14, 2, 0,
      b1, nullptr, hbuf, /*outMode*/1, /*relu*/1, 1568);

  // GEMM4: final conv -> f32 NCHW output. M=768 K=4608 N=1568
  wmma_conv_gemm<1, 4><<<dim3(13, 12), blk, 0, stream>>>(
      Aw2, 4608, 768, nullptr, hbuf, zbuf,
      14, 14, 512, 14, 14, 14, 14, 1, 0,
      b2, out, nullptr, /*outMode*/0, /*relu*/0, 1568);
}
